// BClassifier_19791209300126
// MI455X (gfx1250) — matrix operand order repaired and checked by reading, not compiled, not run on hardware
//
#include <hip/hip_runtime.h>
#include <hip/hip_bf16.h>

// ---------------- types for CDNA5 WMMA ----------------
typedef __attribute__((ext_vector_type(16))) __bf16 v16bf;
typedef __attribute__((ext_vector_type(8)))  __bf16 v8bf;
typedef __attribute__((ext_vector_type(8)))  float  v8f;

#define L_DIM   500
#define LP      512           // padded L
#define D_DIM   128
#define FEAT    1024
#define B_BAGS  32
#define N_INST  4096
#define ROWS    (B_BAGS * N_INST)   // 131072
#define PER_CLS 256
#define EPS_N   1e-8f
#define AP      40            // padded LDS row stride (bf16 elems), 80 B, 16B-aligned

__device__ __forceinline__ unsigned short bfbits(float f) {
  union { float f; unsigned int u; } v; v.f = f;
  unsigned int r = v.u + 0x7FFFu + ((v.u >> 16) & 1u);   // RNE
  return (unsigned short)(r >> 16);
}
__device__ __forceinline__ __bf16 f2bf(float f) {
  return __builtin_bit_cast(__bf16, bfbits(f));
}
// pack two f32 -> packed bf16x2 (one VALU op when v_cvt_pk_bf16_f32 exists)
__device__ __forceinline__ unsigned int pack2(float f0, float f1) {
#if __has_builtin(__builtin_amdgcn_cvt_pk_bf16_f32)
  typedef __attribute__((ext_vector_type(2))) __bf16 v2bf;
  v2bf p = __builtin_amdgcn_cvt_pk_bf16_f32(f0, f1);
  return __builtin_bit_cast(unsigned int, p);
#else
  return ((unsigned int)bfbits(f1) << 16) | (unsigned int)bfbits(f0);
#endif
}
// two 16B LDS loads -> one 16-elem bf16 fragment
__device__ __forceinline__ v16bf ld_frag(const __bf16* p0, const __bf16* p1) {
  v8bf lo = *(const v8bf*)p0;
  v8bf hi = *(const v8bf*)p1;
  return __builtin_shufflevector(lo, hi, 0, 1, 2, 3, 4, 5, 6, 7,
                                 8, 9, 10, 11, 12, 13, 14, 15);
}

// ============================================================
// Kernel 1: H = relu(x @ W1 + b1), H stored [ROWS, 512] (pad=0)
// grid (1024, 4), block 256 (8 waves), bf16 WMMA 16x16x32
// double-buffered LDS; software-pipelined B fragments
// ============================================================
__global__ __launch_bounds__(256) void k_gemm1(
    const float* __restrict__ x, const float* __restrict__ W1,
    const float* __restrict__ b1, float* __restrict__ H)
{
  __shared__ __bf16 As[2][128][AP];
  __shared__ __bf16 Bt[2][128][AP];

  const int tid  = threadIdx.x;
  const int lane = tid & 31;
  const int wave = tid >> 5;
  const int half = lane >> 4;
  const int l16  = lane & 15;
  const int m0   = blockIdx.x * 128;
  const int n0   = blockIdx.y * 128;

  v8f acc[8];
  const v8f vzero = {};
#pragma unroll
  for (int t = 0; t < 8; ++t) acc[t] = vzero;

  auto stage = [&](int buf, int k0) {
    for (int i = tid; i < 128 * 16; i += 256) {
      int r = i >> 4, c2 = (i & 15) * 2;
      const float* px = &x[(size_t)(m0 + r) * FEAT + k0 + c2];
      *(unsigned int*)&As[buf][r][c2] = pack2(px[0], px[1]);
    }
    for (int i = tid; i < 32 * 128; i += 256) {
      int r = i >> 7, c = i & 127;               // r = k-offset, c = col-offset
      int col = n0 + c;
      Bt[buf][c][r] = f2bf(col < L_DIM ? W1[(size_t)(k0 + r) * L_DIM + col] : 0.0f);
    }
  };

  stage(0, 0);
  __syncthreads();

  const int nsteps = FEAT / 32;
  for (int s = 0; s < nsteps; ++s) {
    const int buf = s & 1;
    if (s + 1 < nsteps) stage(buf ^ 1, (s + 1) * 32);

    const int row = wave * 16 + l16;
    v16bf a = ld_frag(&As[buf][row][16 * half], &As[buf][row][16 * half + 8]);
    // software pipeline: load fragment t+1 before consuming fragment t
    v16bf bcur = ld_frag(&Bt[buf][l16][16 * half], &Bt[buf][l16][16 * half + 8]);
#pragma unroll
    for (int t = 0; t < 8; ++t) {
      if (t < 7) {
        const int col = (t + 1) * 16 + l16;
        v16bf bnext = ld_frag(&Bt[buf][col][16 * half], &Bt[buf][col][16 * half + 8]);
        acc[t] = __builtin_amdgcn_wmma_f32_16x16x32_bf16(
            false, a, false, bcur, (short)0, acc[t], false, false);
        bcur = bnext;
      } else {
        acc[t] = __builtin_amdgcn_wmma_f32_16x16x32_bf16(
            false, a, false, bcur, (short)0, acc[t], false, false);
      }
    }
    __syncthreads();
  }

#pragma unroll
  for (int t = 0; t < 8; ++t) {
    const int col = n0 + t * 16 + l16;
    const float bias = (col < L_DIM) ? b1[col] : 0.0f;
#pragma unroll
    for (int r = 0; r < 8; ++r) {
      const int row = m0 + wave * 16 + 8 * half + r;
      float v = acc[t][r] + bias;
      v = v > 0.0f ? v : 0.0f;
      H[(size_t)row * LP + col] = (col < L_DIM) ? v : 0.0f;
    }
  }
}

// ============================================================
// Kernel 2: scores = relu(H @ Wa + ba) @ Va + bv   (fused)
// grid 1024, block 256; K padded to 512 (H pad cols are zero)
// ============================================================
__global__ __launch_bounds__(256) void k_attn(
    const float* __restrict__ H, const float* __restrict__ Wa,
    const float* __restrict__ ba, const float* __restrict__ Va,
    const float* __restrict__ bv, float* __restrict__ scores)
{
  __shared__ __bf16 As[2][128][AP];
  __shared__ __bf16 Bt[2][128][AP];
  __shared__ float  part[128][16];

  const int tid  = threadIdx.x;
  const int lane = tid & 31;
  const int wave = tid >> 5;
  const int half = lane >> 4;
  const int l16  = lane & 15;
  const int m0   = blockIdx.x * 128;

  v8f acc[8];
  const v8f vzero = {};
#pragma unroll
  for (int t = 0; t < 8; ++t) acc[t] = vzero;

  auto stage = [&](int buf, int k0) {
    for (int i = tid; i < 128 * 16; i += 256) {
      int r = i >> 4, c2 = (i & 15) * 2;
      const float* ph = &H[(size_t)(m0 + r) * LP + k0 + c2];
      *(unsigned int*)&As[buf][r][c2] = pack2(ph[0], ph[1]);
    }
    for (int i = tid; i < 32 * 128; i += 256) {
      int r = i >> 7, c = i & 127;
      int krow = k0 + r;
      Bt[buf][c][r] = f2bf(krow < L_DIM ? Wa[(size_t)krow * D_DIM + c] : 0.0f);
    }
  };

  stage(0, 0);
  __syncthreads();

  const int nsteps = LP / 32;
  for (int s = 0; s < nsteps; ++s) {
    const int buf = s & 1;
    if (s + 1 < nsteps) stage(buf ^ 1, (s + 1) * 32);

    const int row = wave * 16 + l16;
    v16bf a = ld_frag(&As[buf][row][16 * half], &As[buf][row][16 * half + 8]);
    v16bf bcur = ld_frag(&Bt[buf][l16][16 * half], &Bt[buf][l16][16 * half + 8]);
#pragma unroll
    for (int t = 0; t < 8; ++t) {
      if (t < 7) {
        const int col = (t + 1) * 16 + l16;
        v16bf bnext = ld_frag(&Bt[buf][col][16 * half], &Bt[buf][col][16 * half + 8]);
        acc[t] = __builtin_amdgcn_wmma_f32_16x16x32_bf16(
            false, a, false, bcur, (short)0, acc[t], false, false);
        bcur = bnext;
      } else {
        acc[t] = __builtin_amdgcn_wmma_f32_16x16x32_bf16(
            false, a, false, bcur, (short)0, acc[t], false, false);
      }
    }
    __syncthreads();
  }

  // relu(.+ba) * Va, deterministic row reduction via LDS
  float prow[8];
#pragma unroll
  for (int r = 0; r < 8; ++r) prow[r] = 0.0f;
#pragma unroll
  for (int t = 0; t < 8; ++t) {
    const int col = t * 16 + l16;
    const float va = Va[col];
    const float bac = ba[col];
#pragma unroll
    for (int r = 0; r < 8; ++r) {
      float v = acc[t][r] + bac;
      v = v > 0.0f ? v : 0.0f;
      prow[r] += v * va;
    }
  }
#pragma unroll
  for (int r = 0; r < 8; ++r)
    part[wave * 16 + 8 * half + r][l16] = prow[r];
  __syncthreads();
  if (tid < 128) {
    float s = 0.0f;
#pragma unroll
    for (int q = 0; q < 16; ++q) s += part[tid][q];
    scores[m0 + tid] = s + bv[0];
  }
}

// ============================================================
// Kernel 3: per-bag softmax over N + bag_feat = A @ H
// grid 32, block 256
// ============================================================
__global__ __launch_bounds__(256) void k_pool(
    const float* __restrict__ scores, const float* __restrict__ H,
    float* __restrict__ bagfeat)
{
  __shared__ float w[N_INST];
  __shared__ float red[256];
  const int b = blockIdx.x, tid = threadIdx.x;
  const float* s = scores + (size_t)b * N_INST;

  float mx = -3.0e38f;
  for (int i = tid; i < N_INST; i += 256) mx = fmaxf(mx, s[i]);
  red[tid] = mx; __syncthreads();
  for (int st = 128; st > 0; st >>= 1) {
    if (tid < st) red[tid] = fmaxf(red[tid], red[tid + st]);
    __syncthreads();
  }
  mx = red[0]; __syncthreads();

  float sum = 0.0f;
  for (int i = tid; i < N_INST; i += 256) {
    float e = __expf(s[i] - mx);
    w[i] = e;
    sum += e;
  }
  red[tid] = sum; __syncthreads();
  for (int st = 128; st > 0; st >>= 1) {
    if (tid < st) red[tid] += red[tid + st];
    __syncthreads();
  }
  const float inv = 1.0f / red[0];
  __syncthreads();

  const float* Hb = H + (size_t)b * N_INST * LP;
  for (int c = tid; c < LP; c += 256) {
    float accv = 0.0f;
    for (int n = 0; n < N_INST; ++n)
      accv = fmaf(w[n], Hb[(size_t)n * LP + c], accv);
    bagfeat[b * LP + c] = accv * inv;
  }
}

// ============================================================
// Kernel 4: class means, norms, logits_mlp, sim, loss, reg
// grid 1, block 512
// ============================================================
__global__ __launch_bounds__(512) void k_small(
    const float* __restrict__ reh, const int* __restrict__ y,
    const float* __restrict__ Wc, const float* __restrict__ bc,
    const float* __restrict__ bagfeat,
    float* __restrict__ clsmean, float* __restrict__ cmn,
    float* __restrict__ out)
{
  __shared__ float red[512];
  __shared__ float sim[64];
  const int tid = threadIdx.x;

  for (int idx = tid; idx < 2 * L_DIM; idx += 512) {
    int c = idx / L_DIM, l = idx - c * L_DIM;
    float sm = 0.0f;
    for (int p = 0; p < PER_CLS; ++p)
      sm += reh[((size_t)c * PER_CLS + p) * L_DIM + l];
    clsmean[c * LP + l] = sm * (1.0f / 256.0f);
  }
  if (tid < 24) { int c = tid / 12, l = L_DIM + tid - c * 12; clsmean[c * LP + l] = 0.0f; }
  __syncthreads();

  const float c0 = clsmean[tid], c1 = clsmean[LP + tid];
  red[tid] = c0 * c0; __syncthreads();
  for (int st = 256; st > 0; st >>= 1) { if (tid < st) red[tid] += red[tid + st]; __syncthreads(); }
  const float d0 = fmaxf(sqrtf(red[0]), EPS_N); __syncthreads();
  red[tid] = c1 * c1; __syncthreads();
  for (int st = 256; st > 0; st >>= 1) { if (tid < st) red[tid] += red[tid + st]; __syncthreads(); }
  const float d1 = fmaxf(sqrtf(red[0]), EPS_N); __syncthreads();

  cmn[tid] = c0 / d0;
  cmn[LP + tid] = c1 / d1;

  red[tid] = (c0 / d0) * (c1 / d1); __syncthreads();
  for (int st = 256; st > 0; st >>= 1) { if (tid < st) red[tid] += red[tid + st]; __syncthreads(); }
  const float regv = 1.0f + red[0];
  __syncthreads();

  if (tid < 64) {
    const int b = tid >> 1, c = tid & 1;
    const float* bf = bagfeat + b * LP;
    float lg = 0.0f, s = 0.0f;
    for (int l = 0; l < L_DIM; ++l) {
      float v = bf[l];
      lg = fmaf(v, Wc[l * 2 + c], lg);
      s  = fmaf(v, clsmean[c * LP + l], s);
    }
    out[tid] = lg + bc[c];
    sim[tid] = s;
  }
  __syncthreads();
  if (tid == 0) {
    float accl = 0.0f;
    for (int b = 0; b < B_BAGS; ++b) {
      int ci = y[b];
      float pos = sim[b * 2 + ci];
      float neg = sim[b * 2 + (1 - ci)];
      accl += log1pf(__expf((neg - pos) * 2.0f));   // 1/TEMP = 2
    }
    out[64] = accl * (1.0f / 32.0f);
    out[65] = regv;
  }
}

// ============================================================
// Kernel 5: copy rehearsal -> out buffer; dists = cos(cm_n, row)
// grid 512 (one per (c,p)), block 256
// ============================================================
__global__ __launch_bounds__(256) void k_dists(
    const float* __restrict__ reh, const float* __restrict__ cmn,
    float* __restrict__ outReh, float* __restrict__ dists)
{
  __shared__ float red[256];
  __shared__ float s_dot;
  const int cp = blockIdx.x;
  const int c = cp >> 8;
  const int tid = threadIdx.x;
  const float* row = reh + (size_t)cp * L_DIM;

  float dot = 0.0f, ss = 0.0f;
  for (int l = tid; l < L_DIM; l += 256) {
    float v = row[l];
    outReh[(size_t)cp * L_DIM + l] = v;
    dot = fmaf(cmn[c * LP + l], v, dot);
    ss  = fmaf(v, v, ss);
  }
  red[tid] = dot; __syncthreads();
  for (int st = 128; st > 0; st >>= 1) { if (tid < st) red[tid] += red[tid + st]; __syncthreads(); }
  if (tid == 0) s_dot = red[0];
  __syncthreads();
  red[tid] = ss; __syncthreads();
  for (int st = 128; st > 0; st >>= 1) { if (tid < st) red[tid] += red[tid + st]; __syncthreads(); }
  if (tid == 0) dists[cp] = s_dot / fmaxf(sqrtf(red[0]), EPS_N);
}

// ============================================================
// Kernel 6: sequential 32-step rehearsal scatter update
// grid 1, block 512 (deterministic tree reductions)
// ============================================================
__global__ __launch_bounds__(512) void k_scan(
    const int* __restrict__ y, const float* __restrict__ bagfeat,
    const float* __restrict__ cmn, float* __restrict__ dists,
    float* __restrict__ outReh)
{
  __shared__ float red[512];
  __shared__ int   redi[256];
  __shared__ float s_d, s_min;
  __shared__ int   s_idx;
  const int tid = threadIdx.x;

  for (int i = 0; i < B_BAGS; ++i) {
    const int ci = y[i];
    const float* bf = bagfeat + i * LP;
    const float v  = (tid < L_DIM) ? bf[tid] : 0.0f;
    const float pd = (tid < L_DIM) ? cmn[ci * LP + tid] * v : 0.0f;

    red[tid] = pd; __syncthreads();
    for (int st = 256; st > 0; st >>= 1) { if (tid < st) red[tid] += red[tid + st]; __syncthreads(); }
    const float dotv = red[0]; __syncthreads();
    red[tid] = v * v; __syncthreads();
    for (int st = 256; st > 0; st >>= 1) { if (tid < st) red[tid] += red[tid + st]; __syncthreads(); }
    if (tid == 0) s_d = dotv / fmaxf(sqrtf(red[0]), EPS_N);
    __syncthreads();

    if (tid < 256) { red[tid] = dists[ci * PER_CLS + tid]; redi[tid] = tid; }
    __syncthreads();
    for (int st = 128; st > 0; st >>= 1) {
      if (tid < st) {
        if (red[tid + st] < red[tid]) { red[tid] = red[tid + st]; redi[tid] = redi[tid + st]; }
      }
      __syncthreads();
    }
    if (tid == 0) { s_min = red[0]; s_idx = redi[0]; }
    __syncthreads();

    if (s_d > s_min) {
      if (tid < L_DIM)
        outReh[((size_t)ci * PER_CLS + s_idx) * L_DIM + tid] = v;
      if (tid == 0) dists[ci * PER_CLS + s_idx] = s_d;
    }
    __syncthreads();
  }
}

// ============================================================
// launch
// ============================================================
extern "C" void kernel_launch(void* const* d_in, const int* in_sizes, int n_in,
                              void* d_out, int out_size, void* d_ws, size_t ws_size,
                              hipStream_t stream) {
  (void)in_sizes; (void)n_in; (void)out_size; (void)ws_size;
  const float* x   = (const float*)d_in[0];
  const int*   y   = (const int*)  d_in[1];
  const float* W1  = (const float*)d_in[2];
  const float* b1  = (const float*)d_in[3];
  const float* Wa  = (const float*)d_in[4];
  const float* ba  = (const float*)d_in[5];
  const float* Va  = (const float*)d_in[6];
  const float* bv  = (const float*)d_in[7];
  const float* Wc  = (const float*)d_in[8];
  const float* bc  = (const float*)d_in[9];
  const float* reh = (const float*)d_in[10];
  float* out = (float*)d_out;
  float* ws  = (float*)d_ws;

  // workspace layout (floats)
  const size_t SZ_H = (size_t)ROWS * LP;          // 67,108,864
  float* H       = ws;
  float* scores  = ws + SZ_H;                     // ROWS
  float* bagfeat = scores + ROWS;                 // 32*512
  float* clsmean = bagfeat + B_BAGS * LP;         // 2*512
  float* cmn     = clsmean + 2 * LP;              // 2*512
  float* dists   = cmn + 2 * LP;                  // 512

  float* outReh = out + 66;                       // [2,256,500]

  k_gemm1<<<dim3(ROWS / 128, LP / 128), 256, 0, stream>>>(x, W1, b1, H);
  k_attn <<<ROWS / 128, 256, 0, stream>>>(H, Wa, ba, Va, bv, scores);
  k_pool <<<B_BAGS, 256, 0, stream>>>(scores, H, bagfeat);
  k_small<<<1, 512, 0, stream>>>(reh, y, Wc, bc, bagfeat, clsmean, cmn, out);
  k_dists<<<2 * PER_CLS, 256, 0, stream>>>(reh, cmn, outReh, dists);
  k_scan <<<1, 512, 0, stream>>>(y, bagfeat, cmn, dists, outReh);
}
